// FixedSlotAttentionMultiHead_56315611185660
// MI455X (gfx1250) — matrix operand order repaired and checked by reading, not compiled, not run on hardware
//
#include <hip/hip_runtime.h>
#include <hip/hip_bf16.h>
#include <stdint.h>

#define B_    16
#define N_    2048
#define D_    256
#define K_    8
#define H_    4
#define U_    64
#define HID_  1024
#define ITERS_ 3
#define M_TOT (B_ * N_)          // 32768 rows
#define SCALE_ 0.0625f           // 256^-0.5
#define LNEPS  1e-5f

typedef __attribute__((ext_vector_type(16))) __bf16 v16bf;
typedef __attribute__((ext_vector_type(8)))  float  v8f;

union Frag16 { v16bf v; uint4 q[2]; };

__device__ __forceinline__ float bf2f(unsigned int u) {
  union { unsigned int i; float f; } c; c.i = (u & 0xFFFFu) << 16; return c.f;
}
__device__ __forceinline__ unsigned short f2bf(float f) {
  union { float f; unsigned int i; } c; c.f = f;
  unsigned int i = c.i;
  return (unsigned short)((i + 0x7FFFu + ((i >> 16) & 1u)) >> 16);
}

// ---------------- prep kernels ----------------

// LayerNorm(embeddings) -> bf16, one row (D=256) per block
__global__ __launch_bounds__(256) void k_ln_emb(const float* __restrict__ x,
                                                const float* __restrict__ g,
                                                const float* __restrict__ b,
                                                unsigned short* __restrict__ o) {
  __shared__ float red[256];
  const int row = blockIdx.x, t = threadIdx.x;
  float v = x[(size_t)row * D_ + t];
  red[t] = v; __syncthreads();
  for (int s = 128; s > 0; s >>= 1) { if (t < s) red[t] += red[t + s]; __syncthreads(); }
  float mean = red[0] * (1.0f / D_); __syncthreads();
  float d = v - mean;
  red[t] = d * d; __syncthreads();
  for (int s = 128; s > 0; s >>= 1) { if (t < s) red[t] += red[t + s]; __syncthreads(); }
  float var = red[0] * (1.0f / D_);
  float y = d * rsqrtf(var + LNEPS) * g[t] + b[t];
  o[(size_t)row * D_ + t] = f2bf(y);
}

// to_keys (mats 0..7) + to_values (mats 8..15) -> bf16 [16][256][256] (row-major e,d)
__global__ __launch_bounds__(256) void k_wcast(const float* __restrict__ tk,
                                               const float* __restrict__ tv,
                                               unsigned short* __restrict__ wb) {
  size_t idx = (size_t)blockIdx.x * 256 + threadIdx.x;
  int mat = (int)(idx >> 16);
  size_t rem = idx & 65535u;
  float v = (mat < 8) ? tk[(size_t)mat * 65536 + rem] : tv[(size_t)(mat - 8) * 65536 + rem];
  wb[idx] = f2bf(v);
}

// diagQ[k][d] = to_queries[k][d][d]
__global__ __launch_bounds__(256) void k_diagq(const float* __restrict__ tq, float* __restrict__ dq) {
  int k = blockIdx.x, d = threadIdx.x;
  dq[k * D_ + d] = tq[(size_t)k * 65536 + (size_t)d * D_ + d];
}

// slots = mu + exp(logsigma)*noise
__global__ __launch_bounds__(256) void k_init_slots(const float* __restrict__ mu,
                                                    const float* __restrict__ ls,
                                                    const float* __restrict__ nz,
                                                    float* __restrict__ slots) {
  int i = blockIdx.x * 256 + threadIdx.x;
  int j = i & (K_ * D_ - 1);
  slots[i] = mu[j] + expf(ls[j]) * nz[i];
}

// ---------------- WMMA projection GEMM ----------------
// Ob[mat][m][d] = sum_e Ab[m][e] * Wb[mat][e][d]   (bf16 in, bf16 out, f32 acc)
__global__ __launch_bounds__(256) void k_gemm_proj(const unsigned short* __restrict__ Ab,
                                                   const unsigned short* __restrict__ Wb,
                                                   unsigned short* __restrict__ Ob) {
  __shared__ unsigned short As[16][264];   // padded rows: 528B stride (bank spread)
  const int mt  = blockIdx.x;              // M tile (16 rows), 0..2047
  const int mat = blockIdx.y;              // 0..15
  const int t   = threadIdx.x;

  { // cooperative load of first 128 cols of the 16x256 bf16 A tile (4KB), 16B per thread
    int r = t >> 4, c = t & 15;
    const uint4* src = (const uint4*)(Ab + (size_t)(mt * 16 + r) * D_) + c;
    *(uint4*)(&As[r][c * 8]) = *src;
    if (mt + 1 < (int)gridDim.x)  // prefetch next M tile's A rows (global_prefetch)
      __builtin_prefetch(Ab + (size_t)((mt + 1) * 16 + r) * D_ + c * 8, 0, 0);
  }
  __syncthreads();

  const int wave = t >> 5, lane = t & 31;
  const int half = lane >> 4, l = lane & 15;
  const int n0 = wave * 32;                 // each wave: two 16-col tiles
  v8f acc0 = {0.f,0.f,0.f,0.f,0.f,0.f,0.f,0.f};
  v8f acc1 = acc0;
  const unsigned short* Wm = Wb + (size_t)mat * 65536;

  for (int k0 = 0; k0 < D_; k0 += 32) {
    Frag16 a, b0, b1;
    // A frag: lane<16 -> M=l, K = k0+{0..7,16..23}; lane>=16 -> K = k0+{8..15,24..31}
    int ao = k0 + (half ? 8 : 0);
    a.q[0] = *(const uint4*)(&As[l][ao]);
    a.q[1] = *(const uint4*)(&As[l][ao + 16]);
    // B frag: lane%16 -> N=n0+l, K order same as A frag (k0+{0..7,16..23} / k0+{8..15,24..31})
    const unsigned short* bcol = Wm + (size_t)(k0 + (half ? 8 : 0)) * D_ + n0 + l;
    #pragma unroll
    for (int s = 0; s < 16; ++s) {
      int kk = (s < 8) ? s : s + 8;
      b0.v[s] = __builtin_bit_cast(__bf16, bcol[(size_t)kk * D_]);
      b1.v[s] = __builtin_bit_cast(__bf16, bcol[(size_t)kk * D_ + 16]);
    }
    acc0 = __builtin_amdgcn_wmma_f32_16x16x32_bf16(false, a.v, false, b0.v, (short)0, acc0, false, false);
    acc1 = __builtin_amdgcn_wmma_f32_16x16x32_bf16(false, a.v, false, b1.v, (short)0, acc1, false, false);
  }

  // C/D: VGPR j -> M = j (+8 for upper lane half), lane%16 -> N
  unsigned short* out = Ob + (size_t)mat * M_TOT * D_ + (size_t)mt * 16 * D_;
  #pragma unroll
  for (int j = 0; j < 8; ++j) {
    int row = j + (half ? 8 : 0);
    out[(size_t)row * D_ + n0 + l]      = f2bf(acc0[j]);
    out[(size_t)row * D_ + n0 + 16 + l] = f2bf(acc1[j]);
  }
}

// ---------------- iteration kernels ----------------

// queries[b,k,d] = LN(slots)[b,k,d]*g+b * diagQ[k,d]; one (b,k) row per block
__global__ __launch_bounds__(256) void k_queries(const float* __restrict__ slots,
                                                 const float* __restrict__ g,
                                                 const float* __restrict__ b,
                                                 const float* __restrict__ dq,
                                                 float* __restrict__ q) {
  __shared__ float red[256];
  const int row = blockIdx.x, t = threadIdx.x, k = row & (K_ - 1);
  float v = slots[(size_t)row * D_ + t];
  red[t] = v; __syncthreads();
  for (int s = 128; s > 0; s >>= 1) { if (t < s) red[t] += red[t + s]; __syncthreads(); }
  float mean = red[0] * (1.0f / D_); __syncthreads();
  float d = v - mean;
  red[t] = d * d; __syncthreads();
  for (int s = 128; s > 0; s >>= 1) { if (t < s) red[t] += red[t + s]; __syncthreads(); }
  float var = red[0] * (1.0f / D_);
  float y = d * rsqrtf(var + LNEPS) * g[t] + b[t];
  q[(size_t)row * D_ + t] = y * dq[k * D_ + t];
}

// dots + softmax over slots + head average -> vis[b,k,n]
// block: (ntile of 64 n, b); thread: h = t&3, nl = t>>2; all K=8 logits in registers
__global__ __launch_bounds__(256) void k_dots_softmax(const unsigned short* __restrict__ keys,
                                                      const float* __restrict__ q,
                                                      float* __restrict__ vis) {
  __shared__ float qs[K_ * D_];
  const int b = blockIdx.y, nt = blockIdx.x, t = threadIdx.x;
  for (int i = t; i < K_ * D_; i += 256) qs[i] = q[(size_t)b * K_ * D_ + i];
  __syncthreads();
  const int h = t & 3, nl = t >> 2, n = nt * 64 + nl;
  float vals[K_];
  #pragma unroll
  for (int k = 0; k < K_; ++k) {
    const uint4* kp = (const uint4*)(keys + ((size_t)k * M_TOT + (size_t)b * N_ + n) * D_ + h * U_);
    const float* qk = &qs[k * D_ + h * U_];
    float acc = 0.f;
    #pragma unroll
    for (int u = 0; u < 8; ++u) {
      uint4 w = kp[u];
      acc += bf2f(w.x) * qk[u*8+0] + bf2f(w.x >> 16) * qk[u*8+1]
           + bf2f(w.y) * qk[u*8+2] + bf2f(w.y >> 16) * qk[u*8+3]
           + bf2f(w.z) * qk[u*8+4] + bf2f(w.z >> 16) * qk[u*8+5]
           + bf2f(w.w) * qk[u*8+6] + bf2f(w.w >> 16) * qk[u*8+7];
    }
    vals[k] = acc * SCALE_;
  }
  float m = vals[0];
  #pragma unroll
  for (int k = 1; k < K_; ++k) m = fmaxf(m, vals[k]);
  float s = 0.f;
  #pragma unroll
  for (int k = 0; k < K_; ++k) { vals[k] = expf(vals[k] - m); s += vals[k]; }
  const float inv = 1.0f / s;
  #pragma unroll
  for (int k = 0; k < K_; ++k) {
    float a = vals[k] * inv;
    a += __shfl_xor(a, 1);       // average over the 4 heads (adjacent lanes, wave32)
    a += __shfl_xor(a, 2);
    a *= 0.25f;
    if (h == 0) vis[((size_t)b * K_ + k) * N_ + n] = a;
  }
}

// renorm over tokens: attn_r[bk,n] = vis[bk,n] / sum_n vis[bk,n]
__global__ __launch_bounds__(256) void k_renorm(const float* __restrict__ vis, float* __restrict__ ar) {
  __shared__ float red[256];
  const int bk = blockIdx.x, t = threadIdx.x;
  const float* src = vis + (size_t)bk * N_;
  float s = 0.f;
  for (int i = t; i < N_; i += 256) s += src[i];
  red[t] = s; __syncthreads();
  for (int st = 128; st > 0; st >>= 1) { if (t < st) red[t] += red[t + st]; __syncthreads(); }
  const float inv = 1.0f / red[0];
  for (int i = t; i < N_; i += 256) ar[(size_t)bk * N_ + i] = src[i] * inv;
}

__global__ __launch_bounds__(256) void k_zero(float* __restrict__ p) {
  p[blockIdx.x * 256 + threadIdx.x] = 0.f;
}

// upd[bk,d] += sum_n values[b,k,n,d] * attn_r[bk,n]   (N split in 8, f32 atomics)
__global__ __launch_bounds__(256) void k_upd(const unsigned short* __restrict__ proj,
                                             const float* __restrict__ ar,
                                             float* __restrict__ upd) {
  __shared__ float aw[256];
  const int split = blockIdx.x, bk = blockIdx.y, t = threadIdx.x;
  const int b = bk >> 3, k = bk & 7, n0 = split * 256;
  aw[t] = ar[(size_t)bk * N_ + n0 + t];
  __syncthreads();
  const unsigned short* vp =
      proj + ((size_t)(K_ + k) * M_TOT + (size_t)b * N_ + n0) * D_ + t;
  float acc = 0.f;
  for (int i = 0; i < 256; ++i) acc += bf2f(vp[(size_t)i * D_]) * aw[i];
  atomicAdd(&upd[(size_t)bk * D_ + t], acc);
}

// fused GRUCell: one (b,k) row per block
__global__ __launch_bounds__(256) void k_gru(const float* __restrict__ upd,
                                             float* __restrict__ slots,
                                             const float* __restrict__ w_ih,
                                             const float* __restrict__ w_hh,
                                             const float* __restrict__ b_ih,
                                             const float* __restrict__ b_hh) {
  __shared__ float xs[256], hs[256];
  const int row = blockIdx.x, t = threadIdx.x;
  xs[t] = upd[(size_t)row * D_ + t];
  hs[t] = slots[(size_t)row * D_ + t];
  __syncthreads();
  float gi[3], gh[3];
  #pragma unroll
  for (int g = 0; g < 3; ++g) {
    const float* wi = w_ih + ((size_t)g * D_ + t) * D_;
    const float* wh = w_hh + ((size_t)g * D_ + t) * D_;
    float a = 0.f, c = 0.f;
    for (int e = 0; e < D_; ++e) { a += wi[e] * xs[e]; c += wh[e] * hs[e]; }
    gi[g] = a + b_ih[g * D_ + t];
    gh[g] = c + b_hh[g * D_ + t];
  }
  float r = 1.0f / (1.0f + expf(-(gi[0] + gh[0])));
  float z = 1.0f / (1.0f + expf(-(gi[1] + gh[1])));
  float n = tanhf(gi[2] + r * gh[2]);
  slots[(size_t)row * D_ + t] = (1.0f - z) * n + z * hs[t];
}

// fused LN + residual MLP: one (b,k) row per block
__global__ __launch_bounds__(256) void k_mlp(float* __restrict__ slots,
                                             const float* __restrict__ g,
                                             const float* __restrict__ b,
                                             const float* __restrict__ w1,
                                             const float* __restrict__ b1,
                                             const float* __restrict__ w2,
                                             const float* __restrict__ b2) {
  __shared__ float red[256];
  __shared__ float pre[256];
  __shared__ float hid[HID_];
  const int row = blockIdx.x, t = threadIdx.x;
  const float s0 = slots[(size_t)row * D_ + t];
  red[t] = s0; __syncthreads();
  for (int s = 128; s > 0; s >>= 1) { if (t < s) red[t] += red[t + s]; __syncthreads(); }
  float mean = red[0] * (1.0f / D_); __syncthreads();
  float d = s0 - mean;
  red[t] = d * d; __syncthreads();
  for (int s = 128; s > 0; s >>= 1) { if (t < s) red[t] += red[t + s]; __syncthreads(); }
  float var = red[0] * (1.0f / D_);
  pre[t] = d * rsqrtf(var + LNEPS) * g[t] + b[t];
  __syncthreads();
  #pragma unroll
  for (int j = 0; j < 4; ++j) {
    const int o = j * 256 + t;
    const float* w = w1 + (size_t)o * D_;
    float acc = 0.f;
    for (int e = 0; e < D_; ++e) acc += w[e] * pre[e];
    hid[o] = fmaxf(acc + b1[o], 0.f);
  }
  __syncthreads();
  const float* w = w2 + (size_t)t * HID_;
  float acc = 0.f;
  for (int i = 0; i < HID_; ++i) acc += w[i] * hid[i];
  slots[(size_t)row * D_ + t] = s0 + acc + b2[t];
}

__global__ __launch_bounds__(256) void k_copy(const float* __restrict__ s, float* __restrict__ o) {
  int i = blockIdx.x * 256 + threadIdx.x;
  o[i] = s[i];
}

// ---------------- launcher ----------------

extern "C" void kernel_launch(void* const* d_in, const int* in_sizes, int n_in,
                              void* d_out, int out_size, void* d_ws, size_t ws_size,
                              hipStream_t stream) {
  const float* embeddings = (const float*)d_in[0];
  const float* noise      = (const float*)d_in[1];
  const float* slots_mu   = (const float*)d_in[2];
  const float* slots_ls   = (const float*)d_in[3];
  const float* to_keys    = (const float*)d_in[4];
  const float* to_queries = (const float*)d_in[5];
  const float* to_values  = (const float*)d_in[6];
  const float* w_ih       = (const float*)d_in[7];
  const float* w_hh       = (const float*)d_in[8];
  const float* b_ih       = (const float*)d_in[9];
  const float* b_hh       = (const float*)d_in[10];
  const float* w1         = (const float*)d_in[11];
  const float* b1         = (const float*)d_in[12];
  const float* w2         = (const float*)d_in[13];
  const float* b2         = (const float*)d_in[14];
  const float* ln_in_g    = (const float*)d_in[15];
  const float* ln_in_b    = (const float*)d_in[16];
  const float* ln_sl_g    = (const float*)d_in[17];
  const float* ln_sl_b    = (const float*)d_in[18];
  const float* ln_ff_g    = (const float*)d_in[19];
  const float* ln_ff_b    = (const float*)d_in[20];

  char* ws = (char*)d_ws;
  unsigned short* emb_bf = (unsigned short*)(ws);                 // 16,777,216 B
  unsigned short* Wb     = (unsigned short*)(ws + 16777216);      //  2,097,152 B
  float*          diagQ  = (float*)(ws + 18874368);               //      8,192 B
  unsigned short* proj   = (unsigned short*)(ws + 18882560);      // 268,435,456 B (keys mats 0-7, values 8-15)
  float*          slots  = (float*)(ws + 287318016);              //    131,072 B
  float*          quer   = (float*)(ws + 287449088);              //    131,072 B
  float*          attn_r = (float*)(ws + 287580160);              //  1,048,576 B
  float*          upd    = (float*)(ws + 288628736);              //    131,072 B

  float* out_slots = (float*)d_out;                 // (B,K,D) = 32768 floats
  float* out_vis   = out_slots + B_ * K_ * D_;      // (B,K,N) = 262144 floats

  // prep
  k_ln_emb<<<M_TOT, 256, 0, stream>>>(embeddings, ln_in_g, ln_in_b, emb_bf);
  k_wcast<<<4096, 256, 0, stream>>>(to_keys, to_values, Wb);
  k_diagq<<<K_, 256, 0, stream>>>(to_queries, diagQ);
  k_init_slots<<<128, 256, 0, stream>>>(slots_mu, slots_ls, noise, slots);

  // key/value projections via WMMA (16 weight matrices)
  k_gemm_proj<<<dim3(M_TOT / 16, 16), 256, 0, stream>>>(emb_bf, Wb, proj);

  // slot-attention iterations
  for (int it = 0; it < ITERS_; ++it) {
    k_queries<<<B_ * K_, 256, 0, stream>>>(slots, ln_sl_g, ln_sl_b, diagQ, quer);
    k_dots_softmax<<<dim3(N_ / 64, B_), 256, 0, stream>>>(proj, quer, out_vis);
    k_renorm<<<B_ * K_, 256, 0, stream>>>(out_vis, attn_r);
    k_zero<<<128, 256, 0, stream>>>(upd);
    k_upd<<<dim3(N_ / 256, B_ * K_), 256, 0, stream>>>(proj, attn_r, upd);
    k_gru<<<B_ * K_, 256, 0, stream>>>(upd, slots, w_ih, w_hh, b_ih, b_hh);
    k_mlp<<<B_ * K_, 256, 0, stream>>>(slots, ln_ff_g, ln_ff_b, w1, b1, w2, b2);
  }
  k_copy<<<128, 256, 0, stream>>>(slots, out_slots);
}
